// MambaDiffuser_57853209477744
// MI455X (gfx1250) — matrix operand order repaired and checked by reading, not compiled, not run on hardware
//
#include <hip/hip_runtime.h>
#include <hip/hip_bf16.h>

// ---- model dims (match reference) ----
#define B_  2
#define S_  512
#define D_  512
#define H_  1024
#define L_  8
#define V_  50257
#define N_  16
#define R_  32
#define NH_ 8
#define DH_ 64
#define M_  (B_*S_)

// GEMM tiling
#define BM 128            // block M tile
#define BN 64             // block N tile
#define BK 32             // K step
#define AST 40            // LDS row stride in halves (80B: conflict-free & 16B aligned)

typedef __attribute__((ext_vector_type(16))) _Float16 v16h;
typedef __attribute__((ext_vector_type(8)))  _Float16 v8h;
typedef __attribute__((ext_vector_type(8)))  float    v8f;

__device__ __forceinline__ float silu_f(float x) { return x / (1.f + expf(-x)); }

// pack two float4 into 8 halves
__device__ __forceinline__ v8h pk8(float4 a, float4 b)
{
    v8h r;
    r[0] = (_Float16)a.x; r[1] = (_Float16)a.y; r[2] = (_Float16)a.z; r[3] = (_Float16)a.w;
    r[4] = (_Float16)b.x; r[5] = (_Float16)b.y; r[6] = (_Float16)b.z; r[7] = (_Float16)b.w;
    return r;
}

// ---------------------------------------------------------------------------
// embed: x[b,s,:] = embed_table[ids[b,s]] + pos[s] + t_embed[timestep[b]]
// ---------------------------------------------------------------------------
__global__ void embed_kernel(const int* __restrict__ ids, const int* __restrict__ ts,
                             const float* __restrict__ emb, const float* __restrict__ pos,
                             const float* __restrict__ temb, float* __restrict__ X)
{
    int idx = blockIdx.x * blockDim.x + threadIdx.x;
    if (idx >= M_ * D_) return;
    int m = idx / D_, d = idx - m * D_;
    int b = m / S_,  s = m - b * S_;
    int tok = ids[b * S_ + s];
    int t   = ts[b];
    X[idx] = emb[(long long)tok * D_ + d] + pos[(long long)s * D_ + d]
           + temb[(long long)t * D_ + d];
}

// ---------------------------------------------------------------------------
// Batched GEMM-NT via WMMA, block-cooperative with LDS double buffering.
//   C = A[M,K] * W[N,K]^T (+bias[n]) (+resid)
// Block: 256 threads = 8 waves -> 128x64 C tile; each wave a 32x32 subtile
// (2x2 fragments, 4 v_wmma per BK=32 step).
// Tiles staged fp32->fp16 through LDS; next tile's global_load_b128s issue
// before current tile's ds_load+wmma (one barrier per K-step).
// K must be a multiple of 32. Rows clamped for loads; stores guarded.
// Fragment layouts per CDNA5 ISA 7.12.2 (wave32):
//   A (16x32 f16): m = lane&15; hf = lane>>4; halves at k = hf*8 + [0..7]
//                  and 16 + hf*8 + [0..7]
//   B (32x16 f16): n = lane&15; halves at k = hf*8 + [0..7], 16 + hf*8 + [0..7]
//   C/D (16x16 f32, 8 VGPRs): m = r + 8*hf, n = lane&15
// ---------------------------------------------------------------------------
__global__ void __launch_bounds__(256)
gemm_nt_kernel(const float* __restrict__ A, int lda, long long sAb, long long sAh,
               const float* __restrict__ W, int ldw, long long sWb, long long sWh,
               float* __restrict__ C, int ldc, long long sCb, long long sCh,
               const float* __restrict__ bias, const float* __restrict__ resid,
               int Mdim, int Ndim, int Kdim, int nh)
{
    __shared__ _Float16 As[2 * BM * AST];   // 20 KB
    __shared__ _Float16 Bs[2 * BN * AST];   // 10 KB

    int z  = blockIdx.z;
    int zb = z / nh, zh = z - zb * nh;
    A += zb * sAb + zh * sAh;
    W += zb * sWb + zh * sWh;
    C += zb * sCb + zh * sCh;
    if (resid) resid += zb * sCb + zh * sCh;

    const int tm = blockIdx.y * BM;
    const int tn = blockIdx.x * BN;
    const int tid  = threadIdx.x;
    const int lane = tid & 31;
    const int w    = tid >> 5;              // wave 0..7
    const int mi   = w >> 1;                // 0..3 (32-row strip)
    const int ni   = w & 1;                 // 0..1 (32-col strip)
    const int hf   = lane >> 4, lm = lane & 15;

    // cooperative-load roles
    const int arow = tid >> 1, acol = (tid & 1) * 16;   // 16 floats of one A row
    const int brow = tid >> 2, bcol = (tid & 3) * 8;    // 8 floats of one W row
    int amg = tm + arow; if (amg >= Mdim) amg = Mdim - 1;
    int bng = tn + brow; if (bng >= Ndim) bng = Ndim - 1;
    const float* Ap = A + (long long)amg * lda + acol;
    const float* Wp = W + (long long)bng * ldw + bcol;

    v8f acc[2][2];
#pragma unroll
    for (int f = 0; f < 2; ++f)
#pragma unroll
        for (int g = 0; g < 2; ++g)
            acc[f][g] = (v8f){0.f,0.f,0.f,0.f,0.f,0.f,0.f,0.f};

    auto g2r = [&](int k0, float4* ar, float4* br) {
        ar[0] = *(const float4*)(Ap + k0);
        ar[1] = *(const float4*)(Ap + k0 + 4);
        ar[2] = *(const float4*)(Ap + k0 + 8);
        ar[3] = *(const float4*)(Ap + k0 + 12);
        br[0] = *(const float4*)(Wp + k0);
        br[1] = *(const float4*)(Wp + k0 + 4);
    };
    auto r2l = [&](int buf, const float4* ar, const float4* br) {
        _Float16* Ad = As + buf * (BM * AST) + arow * AST + acol;
        *(v8h*)(Ad)     = pk8(ar[0], ar[1]);
        *(v8h*)(Ad + 8) = pk8(ar[2], ar[3]);
        _Float16* Bd = Bs + buf * (BN * AST) + brow * AST + bcol;
        *(v8h*)(Bd) = pk8(br[0], br[1]);
    };

    const int ko = hf * 8;    // A fragment k-offset
    const int kb = hf * 8;    // B fragment k-offset
    const int nk = Kdim >> 5;

    float4 ar[4], br[2];
    g2r(0, ar, br);
    r2l(0, ar, br);
    __syncthreads();

    for (int kt = 0; kt < nk; ++kt) {
        float4 nar[4], nbr[2];
        const bool more = (kt + 1) < nk;
        if (more) g2r((kt + 1) << 5, nar, nbr);     // issue next global loads early

        // compute on buffer kt&1
        const _Float16* Ab = As + (kt & 1) * (BM * AST);
        const _Float16* Bb = Bs + (kt & 1) * (BN * AST);
        v16h af[2], bf[2];
#pragma unroll
        for (int f = 0; f < 2; ++f) {
            const _Float16* p = Ab + (mi * 32 + f * 16 + lm) * AST + ko;
            v8h lo = *(const v8h*)(p);
            v8h hi = *(const v8h*)(p + 16);
            af[f] = __builtin_shufflevector(lo, hi, 0,1,2,3,4,5,6,7,8,9,10,11,12,13,14,15);
        }
#pragma unroll
        for (int g = 0; g < 2; ++g) {
            const _Float16* p = Bb + (ni * 32 + g * 16 + lm) * AST + kb;
            v8h lo = *(const v8h*)(p);
            v8h hi = *(const v8h*)(p + 16);
            bf[g] = __builtin_shufflevector(lo, hi, 0,1,2,3,4,5,6,7,8,9,10,11,12,13,14,15);
        }
#pragma unroll
        for (int f = 0; f < 2; ++f)
#pragma unroll
            for (int g = 0; g < 2; ++g)
                acc[f][g] = __builtin_amdgcn_wmma_f32_16x16x32_f16(
                    false, af[f], false, bf[g], (short)0, acc[f][g], false, false);

        if (more) r2l((kt + 1) & 1, nar, nbr);      // stage next tile into other buffer
        __syncthreads();                            // covers read-done + write-visible
    }

    // epilogue: wave writes its 32x32 region
#pragma unroll
    for (int g = 0; g < 2; ++g) {
        const int cn = tn + ni * 32 + g * 16 + lm;
        if (cn < Ndim) {
            const float bv = bias ? bias[cn] : 0.f;
#pragma unroll
            for (int f = 0; f < 2; ++f) {
#pragma unroll
                for (int r = 0; r < 8; ++r) {
                    const int cm = tm + mi * 32 + f * 16 + r + 8 * hf;
                    if (cm < Mdim) {
                        long long ci = (long long)cm * ldc + cn;
                        float v = acc[f][g][r] + bv;
                        if (resid) v += resid[ci];
                        C[ci] = v;
                    }
                }
            }
        }
    }
}

// ---------------------------------------------------------------------------
// V transpose: VT[b, hd, d, s] = QKV[b, s, 2D + hd*DH + d]
// ---------------------------------------------------------------------------
__global__ void vtrans_kernel(const float* __restrict__ qkv, float* __restrict__ vt)
{
    int idx = blockIdx.x * blockDim.x + threadIdx.x;
    if (idx >= B_ * NH_ * DH_ * S_) return;
    int s   = idx & (S_ - 1);
    int r   = idx >> 9;            // /S_
    int d   = r & (DH_ - 1);
    r >>= 6;                       // /DH_
    int hd  = r & (NH_ - 1);
    int b   = r >> 3;              // /NH_
    vt[idx] = qkv[((long long)(b * S_ + s)) * (3 * D_) + 2 * D_ + hd * DH_ + d];
}

// ---------------------------------------------------------------------------
// depthwise conv1d (k=3, pad 1) over sequence + SiLU, on first H cols of xz
// ---------------------------------------------------------------------------
__global__ void dwconv_silu_kernel(const float* __restrict__ xz,
                                   const float* __restrict__ cw,
                                   const float* __restrict__ cb,
                                   float* __restrict__ out)
{
    int idx = blockIdx.x * blockDim.x + threadIdx.x;
    if (idx >= M_ * H_) return;
    int m = idx / H_, h = idx - m * H_;
    int b = m / S_,  s = m - b * S_;
    float acc = cb[h];
#pragma unroll
    for (int j = 0; j < 3; ++j) {
        int t = s - 1 + j;
        if (t >= 0 && t < S_)
            acc += xz[((long long)(b * S_ + t)) * (2 * H_) + h] * cw[h * 3 + j];
    }
    out[idx] = silu_f(acc);
}

__global__ void softplus_kernel(float* __restrict__ x, int n)
{
    int i = blockIdx.x * blockDim.x + threadIdx.x;
    if (i >= n) return;
    float v = x[i];
    x[i] = (v > 20.f) ? v : log1pf(expf(v));
}

// tmp = y * silu(z), z = second half of xz
__global__ void gate_mul_kernel(const float* __restrict__ y,
                                const float* __restrict__ xz,
                                float* __restrict__ out)
{
    int idx = blockIdx.x * blockDim.x + threadIdx.x;
    if (idx >= M_ * H_) return;
    int m = idx / H_, h = idx - m * H_;
    float z = xz[(long long)m * (2 * H_) + H_ + h];
    out[idx] = y[idx] * silu_f(z);
}

__global__ void gelu_kernel(float* __restrict__ x, int n)
{
    int i = blockIdx.x * blockDim.x + threadIdx.x;
    if (i >= n) return;
    float v = x[i];
    x[i] = 0.5f * v * (1.f + erff(v * 0.70710678118654752f));
}

// ---------------------------------------------------------------------------
// selective-scan: one thread per (b,h); N=16 states in registers, seq loop
// ---------------------------------------------------------------------------
__global__ void ssm_kernel(const float* __restrict__ xm, const float* __restrict__ dt,
                           const float* __restrict__ xd, const float* __restrict__ A_log,
                           const float* __restrict__ Dp, float* __restrict__ y)
{
    int idx = blockIdx.x * blockDim.x + threadIdx.x;
    if (idx >= B_ * H_) return;
    int b = idx / H_, h = idx - b * H_;

    float An[N_], st[N_];
#pragma unroll
    for (int n = 0; n < N_; ++n) { An[n] = -expf(A_log[n]); st[n] = 0.f; }
    float Dh = Dp[h];

    for (int s = 0; s < S_; ++s) {
        int m = b * S_ + s;
        float u = xm[(long long)m * H_ + h];
        float d = dt[(long long)m * H_ + h];
        const float* xdr = xd + (long long)m * (R_ + 2 * N_);
        float yv = 0.f;
#pragma unroll
        for (int n = 0; n < N_; ++n) {
            float Ae = expf(d * An[n]);
            st[n] = Ae * st[n] + xdr[R_ + n] * u;
            yv   += st[n] * xdr[R_ + N_ + n];
        }
        y[(long long)m * H_ + h] = yv + u * Dh;
    }
}

// ---------------------------------------------------------------------------
// layernorm over last dim (Ddim), one 256-thread block per row
// ---------------------------------------------------------------------------
__global__ void ln_kernel(const float* __restrict__ x, const float* __restrict__ g,
                          const float* __restrict__ b, float* __restrict__ out, int Ddim)
{
    long long row = blockIdx.x;
    const float* p = x + row * Ddim;
    float* q = out + row * Ddim;
    __shared__ float s1[256], s2[256];
    int tid = threadIdx.x;
    float a = 0.f, sq = 0.f;
    for (int i = tid; i < Ddim; i += 256) { float v = p[i]; a += v; sq += v * v; }
    s1[tid] = a; s2[tid] = sq; __syncthreads();
    for (int off = 128; off > 0; off >>= 1) {
        if (tid < off) { s1[tid] += s1[tid + off]; s2[tid] += s2[tid + off]; }
        __syncthreads();
    }
    float mean = s1[0] / Ddim;
    float var  = s2[0] / Ddim - mean * mean;
    float inv  = rsqrtf(var + 1e-5f);
    for (int i = tid; i < Ddim; i += 256)
        q[i] = (p[i] - mean) * inv * g[i] + b[i];
}

// ---------------------------------------------------------------------------
// row softmax with scale, one 256-thread block per row of length `len`
// ---------------------------------------------------------------------------
__global__ void softmax_kernel(float* __restrict__ x, int len, float scale)
{
    long long row = blockIdx.x;
    float* p = x + row * len;
    __shared__ float red[256];
    int tid = threadIdx.x;
    float mx = -1e30f;
    for (int i = tid; i < len; i += 256) mx = fmaxf(mx, p[i] * scale);
    red[tid] = mx; __syncthreads();
    for (int off = 128; off > 0; off >>= 1) {
        if (tid < off) red[tid] = fmaxf(red[tid], red[tid + off]);
        __syncthreads();
    }
    mx = red[0]; __syncthreads();
    float sm = 0.f;
    for (int i = tid; i < len; i += 256) { float e = expf(p[i] * scale - mx); p[i] = e; sm += e; }
    red[tid] = sm; __syncthreads();
    for (int off = 128; off > 0; off >>= 1) {
        if (tid < off) red[tid] += red[tid + off];
        __syncthreads();
    }
    float inv = 1.f / red[0];
    for (int i = tid; i < len; i += 256) p[i] *= inv;
}

// ---------------------------------------------------------------------------
// host-side helpers / orchestration
// ---------------------------------------------------------------------------
static inline int cdiv(int a, int b) { return (a + b - 1) / b; }

static void gemm_nt(hipStream_t st, const float* A, int lda, const float* W, int ldw,
                    float* C, int ldc, const float* bias, const float* resid,
                    int M, int Nn, int K, int batch = 1, int nh = 1,
                    long long sAb = 0, long long sAh = 0,
                    long long sWb = 0, long long sWh = 0,
                    long long sCb = 0, long long sCh = 0)
{
    dim3 g(cdiv(Nn, BN), cdiv(M, BM), batch);
    gemm_nt_kernel<<<g, 256, 0, st>>>(A, lda, sAb, sAh, W, ldw, sWb, sWh,
                                      C, ldc, sCb, sCh, bias, resid, M, Nn, K, nh);
}

extern "C" void kernel_launch(void* const* d_in, const int* in_sizes, int n_in,
                              void* d_out, int out_size, void* d_ws, size_t ws_size,
                              hipStream_t stream)
{
    const int*   ids     = (const int*)  d_in[0];
    const int*   ts      = (const int*)  d_in[1];
    const float* emb     = (const float*)d_in[2];
    const float* pos     = (const float*)d_in[3];
    const float* temb    = (const float*)d_in[4];
    const float* m_in_w  = (const float*)d_in[5];
    const float* m_cw    = (const float*)d_in[6];
    const float* m_cb    = (const float*)d_in[7];
    const float* m_xp_w  = (const float*)d_in[8];
    const float* m_dt_w  = (const float*)d_in[9];
    const float* m_dt_b  = (const float*)d_in[10];
    const float* m_Alog  = (const float*)d_in[11];
    const float* m_Dp    = (const float*)d_in[12];
    const float* m_out_w = (const float*)d_in[13];
    const float* ln1_g   = (const float*)d_in[14];
    const float* ln1_b   = (const float*)d_in[15];
    const float* ln2_g   = (const float*)d_in[16];
    const float* ln2_b   = (const float*)d_in[17];
    const float* ai_w    = (const float*)d_in[18];
    const float* ai_b    = (const float*)d_in[19];
    const float* ao_w    = (const float*)d_in[20];
    const float* ao_b    = (const float*)d_in[21];
    const float* f_w1    = (const float*)d_in[22];
    const float* f_b1    = (const float*)d_in[23];
    const float* f_w2    = (const float*)d_in[24];
    const float* f_b2    = (const float*)d_in[25];
    const float* out_w   = (const float*)d_in[26];
    const float* out_b   = (const float*)d_in[27];
    float* logits = (float*)d_out;

    // workspace carve-up (fp32 elements), ~65 MB total
    float* ws = (float*)d_ws;
    size_t o = 0;
    float* X    = ws + o; o += (size_t)M_ * D_;          // residual stream
    float* XZ   = ws + o; o += (size_t)M_ * 2 * H_;      // mamba in-proj
    float* XM   = ws + o; o += (size_t)M_ * H_;          // conv+silu out
    float* XD   = ws + o; o += (size_t)M_ * (R_ + 2*N_); // x-proj out
    float* DT   = ws + o; o += (size_t)M_ * H_;          // softplus(dt)
    float* Y    = ws + o; o += (size_t)M_ * H_;          // ssm out
    float* TMP  = ws + o; o += (size_t)M_ * H_;          // y*silu(z)
    float* HB   = ws + o; o += (size_t)M_ * D_;          // ln out
    float* QKV  = ws + o; o += (size_t)M_ * 3 * D_;
    float* SC   = ws + o; o += (size_t)B_ * NH_ * S_ * S_;
    float* VT   = ws + o; o += (size_t)B_ * NH_ * DH_ * S_; // transposed V
    float* OB   = ws + o; o += (size_t)M_ * D_;
    float* F1   = ws + o; o += (size_t)M_ * 4 * D_;
    (void)ws_size; (void)n_in; (void)in_sizes; (void)out_size;

    embed_kernel<<<cdiv(M_ * D_, 256), 256, 0, stream>>>(ids, ts, emb, pos, temb, X);

    for (int l = 0; l < L_; ++l) {
        // ---------------- Mamba block ----------------
        gemm_nt(stream, X, D_, m_in_w + (size_t)l * 2 * H_ * D_, D_,
                XZ, 2 * H_, nullptr, nullptr, M_, 2 * H_, D_);
        dwconv_silu_kernel<<<cdiv(M_ * H_, 256), 256, 0, stream>>>(
            XZ, m_cw + (size_t)l * H_ * 3, m_cb + (size_t)l * H_, XM);
        gemm_nt(stream, XM, H_, m_xp_w + (size_t)l * (R_ + 2*N_) * H_, H_,
                XD, R_ + 2 * N_, nullptr, nullptr, M_, R_ + 2 * N_, H_);
        gemm_nt(stream, XD, R_ + 2 * N_, m_dt_w + (size_t)l * H_ * R_, R_,
                DT, H_, m_dt_b + (size_t)l * H_, nullptr, M_, H_, R_);
        softplus_kernel<<<cdiv(M_ * H_, 256), 256, 0, stream>>>(DT, M_ * H_);
        ssm_kernel<<<cdiv(B_ * H_, 256), 256, 0, stream>>>(
            XM, DT, XD, m_Alog + (size_t)l * N_, m_Dp + (size_t)l * H_, Y);
        gate_mul_kernel<<<cdiv(M_ * H_, 256), 256, 0, stream>>>(Y, XZ, TMP);
        gemm_nt(stream, TMP, H_, m_out_w + (size_t)l * D_ * H_, H_,
                X, D_, nullptr, /*resid=*/X, M_, D_, H_);

        // ---------------- Transformer block ----------------
        ln_kernel<<<M_, 256, 0, stream>>>(X, ln1_g + (size_t)l * D_, ln1_b + (size_t)l * D_, HB, D_);
        gemm_nt(stream, HB, D_, ai_w + (size_t)l * 3 * D_ * D_, D_,
                QKV, 3 * D_, ai_b + (size_t)l * 3 * D_, nullptr, M_, 3 * D_, D_);

        // scores[b,h] = q @ k^T  (batched over B*NH)
        gemm_nt(stream, QKV, 3 * D_, QKV + D_, 3 * D_, SC, S_,
                nullptr, nullptr, S_, S_, DH_, B_ * NH_, NH_,
                (long long)S_ * 3 * D_, DH_,
                (long long)S_ * 3 * D_, DH_,
                (long long)NH_ * S_ * S_, (long long)S_ * S_);
        softmax_kernel<<<B_ * NH_ * S_, 256, 0, stream>>>(SC, S_, 0.125f /* 1/sqrt(64) */);

        // o[b,h] = att @ v  -> transpose V, then NT GEMM
        vtrans_kernel<<<cdiv(B_ * NH_ * DH_ * S_, 256), 256, 0, stream>>>(QKV, VT);
        gemm_nt(stream, SC, S_, VT, S_, OB, D_, nullptr, nullptr,
                S_, DH_, S_, B_ * NH_, NH_,
                (long long)NH_ * S_ * S_, (long long)S_ * S_,
                (long long)NH_ * DH_ * S_, (long long)DH_ * S_,
                (long long)S_ * D_, DH_);

        gemm_nt(stream, OB, D_, ao_w + (size_t)l * D_ * D_, D_,
                X, D_, ao_b + (size_t)l * D_, /*resid=*/X, M_, D_, D_);

        ln_kernel<<<M_, 256, 0, stream>>>(X, ln2_g + (size_t)l * D_, ln2_b + (size_t)l * D_, HB, D_);
        gemm_nt(stream, HB, D_, f_w1 + (size_t)l * 4 * D_ * D_, D_,
                F1, 4 * D_, f_b1 + (size_t)l * 4 * D_, nullptr, M_, 4 * D_, D_);
        gelu_kernel<<<cdiv(M_ * 4 * D_, 256), 256, 0, stream>>>(F1, M_ * 4 * D_);
        gemm_nt(stream, F1, 4 * D_, f_w2 + (size_t)l * D_ * 4 * D_, 4 * D_,
                X, D_, f_b2 + (size_t)l * D_, /*resid=*/X, M_, D_, 4 * D_);
    }

    // final vocab projection: logits = X @ out_w^T + out_b  (N=50257 ragged)
    gemm_nt(stream, X, D_, out_w, D_, logits, V_, out_b, nullptr, M_, V_, D_);
}
